// DySnakeConv_58454504898709
// MI455X (gfx1250) — matrix operand order repaired and checked by reading, not compiled, not run on hardware
//
#include <hip/hip_runtime.h>
#include <math.h>

typedef _Float16 h8  __attribute__((ext_vector_type(8)));
typedef _Float16 h16 __attribute__((ext_vector_type(16)));
typedef float    f8  __attribute__((ext_vector_type(8)));

#define DSROW 584   // padded LDS row (halfwords) for 576 reduction values, keeps 16B align
#define PW    136   // padded spatial dim (128 + 2*4 halo)
#define PP    (PW * PW)

__device__ __forceinline__ int iclamp(int v, int lo, int hi) {
    return v < lo ? lo : (v > hi ? hi : v);
}

__device__ __forceinline__ h16 hcat(h8 lo, h8 hi) {
    return __builtin_shufflevector(lo, hi, 0,1,2,3,4,5,6,7,8,9,10,11,12,13,14,15);
}

// ---------------------------------------------------------------------------
// Pack kernels
// ---------------------------------------------------------------------------

// x (2,64,128,128) f32 NCHW -> xhp f16 NHWC with 4-pixel zero halo:
// [b][y+4][h+4][c], dims (2,136,136,64). Kills all boundary masking in GEMMs.
__global__ __launch_bounds__(256) void kpackxpad(const float* __restrict__ x,
                                                 _Float16* __restrict__ xhp) {
    int i = blockIdx.x * 256 + threadIdx.x;          // 2*136*136*64
    if (i >= 2 * PP * 64) return;
    int c   = i & 63;
    int p   = i >> 6;
    int col = p % PW;
    int q   = p / PW;
    int row = q % PW;
    int b   = q / PW;
    int r = row - 4, cc = col - 4;
    float v = 0.f;
    if ((unsigned)r < 128u && (unsigned)cc < 128u)
        v = x[((b * 64 + c) << 14) + (r << 7) + cc];
    xhp[i] = (_Float16)v;
}

// conv0_w (64,64,9,9) -> WMMA-B layout [tap81][kc2][oc64][c32] f16
__global__ __launch_bounds__(256) void kpackw0(const float* __restrict__ w,
                                               _Float16* __restrict__ wB) {
    int i = blockIdx.x * 256 + threadIdx.x;          // 331776
    if (i >= 331776) return;
    int oc  = (i >> 5) & 63;
    int cl  = i & 31;
    int kc  = (i >> 11) & 1;
    int tap = i >> 12;
    int c   = kc * 32 + cl;
    int ky  = tap / 9, kx = tap % 9;
    wB[i] = (_Float16)w[((oc * 64 + c) * 9 + ky) * 9 + kx];
}

// dscx_w/dscy_w (64,64,9) -> WMMA-B layout [br2][chunk18][oc64][c32] f16, kc = kk*64+c
__global__ __launch_bounds__(256) void kpackdw(const float* __restrict__ wx,
                                               const float* __restrict__ wy,
                                               _Float16* __restrict__ wB) {
    int i = blockIdx.x * 256 + threadIdx.x;          // 73728
    if (i >= 73728) return;
    int oc    = (i >> 5) & 63;
    int cl    = i & 31;
    int chunk = (i >> 11) % 18;
    int br    = i / 36864;
    int kc    = chunk * 32 + cl;
    int c     = kc & 63, kk = kc >> 6;
    const float* w = br ? wy : wx;
    wB[i] = (_Float16)w[(oc * 64 + c) * 9 + kk];
}

// offset-conv weights -> WMMA-B [br2][chunk18][oc16][c32] f16.
// chunk = tap*2 + chalf (taps of the 3x3). oc columns 9..15 are zero padding.
// Branch 0 uses offx_w rows 0..8 (y-offsets); branch 1 uses offy_w rows 9..17.
__global__ __launch_bounds__(256) void kpackow(const float* __restrict__ wx,
                                               const float* __restrict__ wy,
                                               _Float16* __restrict__ wB) {
    int i = blockIdx.x * 256 + threadIdx.x;          // 2*18*32*16 = 18432
    if (i >= 18432) return;
    int oc    = (i >> 5) & 15;
    int cl    = i & 31;
    int chunk = (i >> 9) % 18;
    int br    = i / 9216;
    int tap   = chunk >> 1, chalf = chunk & 1;
    int c     = chalf * 32 + cl;
    int ky    = tap / 3, kx = tap % 3;
    float v = 0.f;
    if (oc < 9) {
        int ocw = br ? (9 + oc) : oc;
        const float* w = br ? wy : wx;
        v = w[((ocw * 64 + c) * 3 + ky) * 3 + kx];   // (18,64,3,3)
    }
    wB[i] = (_Float16)v;
}

// ---------------------------------------------------------------------------
// conv0: implicit GEMM 9x9 conv, 64->64 ch, WMMA f16->f32, fused BN+SiLU.
// Block = 128 threads (4 waves), tile = 16 pixels (contiguous h) x 64 oc.
// A = input pixels from the zero-padded NHWC f16 buffer (no masking needed):
//   per-lane A = two contiguous 8-f16 runs (ISA 16-bit A 16x32 layout).
// B = packed weights, per-lane (oc=lane&15) 16 c values in two 8-runs, same k order as A.
// ---------------------------------------------------------------------------
__global__ __launch_bounds__(128) void kconv0(const _Float16* __restrict__ xhp,
                                              const _Float16* __restrict__ w0,
                                              const float* __restrict__ g,
                                              const float* __restrict__ bb,
                                              const float* __restrict__ mm,
                                              const float* __restrict__ vv,
                                              float* __restrict__ out) {
    __shared__ __align__(16) float ldsC[64 * 16];
    int bid  = blockIdx.x;                           // 2048 blocks
    int h0   = (bid & 7) << 4;
    int y    = (bid >> 3) & 127;
    int b    = bid >> 10;
    int tid  = threadIdx.x;
    int lane = tid & 31, og = tid >> 5;
    int m    = lane & 15, s = lane >> 4;

    // per-lane base into padded buffer at (y, h0+m), halo origin already folded
    const _Float16* abase = xhp + ((size_t)(b * PW + y) * PW + (h0 + m)) * 64 + s * 8;
    f8 acc = {};
    for (int dy = 0; dy < 9; ++dy) {
        const _Float16* arow = abase + (size_t)dy * PW * 64;
        for (int dx = 0; dx < 9; ++dx) {
            int tap = dy * 9 + dx;
            const _Float16* ap = arow + dx * 64;
            const _Float16* wp = w0 + tap * 4096 + (og * 16 + m) * 32 + s * 8;
#pragma unroll
            for (int kc = 0; kc < 2; ++kc) {
                h16 a  = hcat(*(const h8*)(ap + kc * 32),
                              *(const h8*)(ap + kc * 32 + 16));
                const _Float16* wq = wp + kc * 2048;
                h16 bm = hcat(*(const h8*)wq, *(const h8*)(wq + 16));
                acc = __builtin_amdgcn_wmma_f32_16x16x32_f16(
                          false, a, false, bm, (short)0, acc, false, false);
            }
        }
    }
    // D layout: VGPR r -> pix = r + 8*s, N = oc' = lane&15. Transpose via LDS.
#pragma unroll
    for (int r = 0; r < 8; ++r)
        ldsC[(og * 16 + m) * 16 + (r + 8 * s)] = acc[r];
    __syncthreads();
    int oc = tid & 63, half = tid >> 6;
    float sc = g[oc] * rsqrtf(vv[oc] + 1e-5f);
    float sh = bb[oc] - mm[oc] * sc;
    float* op = out + (((b * 192 + oc) << 14) + (y << 7) + h0 + half * 8);
#pragma unroll
    for (int i = 0; i < 8; ++i) {
        float z = ldsC[oc * 16 + half * 8 + i] * sc + sh;
        op[i] = z / (1.f + expf(-z));                 // SiLU
    }
}

// ---------------------------------------------------------------------------
// Offset conv as WMMA implicit GEMM: 3x3, 64 ch -> 9 channels per branch
// (N=16 tile, columns 9..15 hit zero weights). K = 9 taps * 64 ch = 18 chunks.
// 4 waves/block: og -> (branch = og&1, K-half = og>>1); partial sums combined
// in LDS; fused bias + BN + tanh epilogue -> offbuf[br][b][k][w][h] f32.
// ---------------------------------------------------------------------------
__global__ __launch_bounds__(128) void koffw(const _Float16* __restrict__ xhp,
                                             const _Float16* __restrict__ owB,
    const float* __restrict__ bxw, const float* __restrict__ gxw,
    const float* __restrict__ bbxw, const float* __restrict__ mxw,
    const float* __restrict__ vxw,
    const float* __restrict__ byw, const float* __restrict__ gyw,
    const float* __restrict__ bbyw, const float* __restrict__ myw,
    const float* __restrict__ vyw,
    float* __restrict__ offbuf) {
    __shared__ __align__(16) float ldsC[4 * 16 * 16];
    int bid  = blockIdx.x;                           // 2048 blocks
    int h0   = (bid & 7) << 4;
    int y    = (bid >> 3) & 127;
    int b    = bid >> 10;
    int tid  = threadIdx.x;
    int lane = tid & 31, og = tid >> 5;
    int m    = lane & 15, s = lane >> 4;
    int br   = og & 1, kh = og >> 1;

    // pad-1 conv inside a pad-4 buffer: origin shift = +3
    const _Float16* abase = xhp + ((size_t)(b * PW + y + 3) * PW + (h0 + m + 3)) * 64 + s * 8;
    f8 acc = {};
#pragma unroll
    for (int cnk = 0; cnk < 9; ++cnk) {
        int chunk = kh * 9 + cnk;
        int tap   = chunk >> 1, chalf = chunk & 1;
        int dy    = tap / 3, dx = tap % 3;
        const _Float16* ap = abase + ((size_t)dy * PW + dx) * 64 + chalf * 32;
        h16 a  = hcat(*(const h8*)ap, *(const h8*)(ap + 16));
        const _Float16* wq = owB + ((br * 18 + chunk) * 16 + m) * 32 + s * 8;
        h16 bm = hcat(*(const h8*)wq, *(const h8*)(wq + 16));
        acc = __builtin_amdgcn_wmma_f32_16x16x32_f16(
                  false, a, false, bm, (short)0, acc, false, false);
    }
#pragma unroll
    for (int r = 0; r < 8; ++r)
        ldsC[og * 256 + m * 16 + (r + 8 * s)] = acc[r];
    __syncthreads();
    // combine K-halves, apply bias+BN+tanh for the 9 real channels
    for (int task = tid; task < 288; task += 128) {
        int pix = task & 15;
        int n   = (task >> 4) % 9;
        int brt = task / 144;
        float v = ldsC[brt * 256 + n * 16 + pix] + ldsC[(brt + 2) * 256 + n * 16 + pix];
        int ocw = brt ? (9 + n) : n;
        float bias = brt ? byw[ocw]  : bxw[ocw];
        float gg   = brt ? gyw[ocw]  : gxw[ocw];
        float bbb  = brt ? bbyw[ocw] : bbxw[ocw];
        float mmm  = brt ? myw[ocw]  : mxw[ocw];
        float vvv  = brt ? vyw[ocw]  : vxw[ocw];
        float sc = gg * rsqrtf(vvv + 1e-5f);
        offbuf[(((brt * 2 + b) * 9 + n) << 14) + (y << 7) + h0 + pix] =
            tanhf((v + bias) * sc + (bbb - mmm * sc));
    }
}

// ---------------------------------------------------------------------------
// Fused snake sampling + deformable conv GEMM (K = 9*64 = 576).
// Block = 128 threads, tile = (br,b,w, 16 pixels along h).
// Phase 1: snake cumsum + bilinear-sample batch-0 features into LDS (f16).
// Phase 2: WMMA over 18 chunks of 32; A from LDS (pixels), B = packed weights.
// Epilogue: +bias, transpose via LDS, store preGN f32.
// ---------------------------------------------------------------------------
__global__ __launch_bounds__(128) void kdsc(const _Float16* __restrict__ xhp,
                                            const float* __restrict__ offbuf,
                                            const _Float16* __restrict__ dwB,
                                            const float* __restrict__ dbx,
                                            const float* __restrict__ dby,
                                            float* __restrict__ preGN) {
    __shared__ __align__(16) _Float16 ldsS[16 * DSROW];
    __shared__ __align__(16) float    ldsC[64 * 16];
    int bid = blockIdx.x;                            // 4096 blocks
    int h0  = (bid & 7) << 4;
    int w   = (bid >> 3) & 127;
    int b   = (bid >> 10) & 1;
    int br  = bid >> 11;
    int tid = threadIdx.x;

    const float* offp = offbuf + (((br * 2 + b) * 9) << 14) + (w << 7);

    for (int task = tid; task < 144; task += 128) {
        int pix = task & 15;
        int k   = task >> 4;                          // 0..8
        int h   = h0 + pix;
        float cum = 0.f;
        if (k > 4)      { for (int j = 5; j <= k; ++j) cum += offp[(j << 14) + h]; }
        else if (k < 4) { for (int j = k; j <= 3; ++j) cum += offp[(j << 14) + h]; }
        float yf, xf;
        if (br == 0) { yf = (float)w + cum;       xf = (float)(h + k - 4); }
        else         { yf = (float)(w + k - 4);   xf = (float)h + cum;     }
        int y0 = (int)floorf(yf), x0 = (int)floorf(xf);
        int y1 = y0 + 1,          x1 = x0 + 1;
        int y0c = iclamp(y0, 0, 127), y1c = iclamp(y1, 0, 127);
        int x0c = iclamp(x0, 0, 127), x1c = iclamp(x1, 0, 127);
        float y0f = (float)iclamp(y0, 0, 128), y1f = (float)iclamp(y1, 0, 128);
        float x0f = (float)iclamp(x0, 0, 128), x1f = (float)iclamp(x1, 0, 128);
        float wa0 = (y1f - yf) * (x1f - xf);
        float wc0 = (y1f - yf) * (xf - x0f);
        float wa1 = (yf - y0f) * (x1f - xf);
        float wc1 = (yf - y0f) * (xf - x0f);
        // batch-0 features in padded buffer: +4 halo shift
        const _Float16* p00 = xhp + ((size_t)(y0c + 4) * PW + (x0c + 4)) * 64;
        const _Float16* p01 = xhp + ((size_t)(y0c + 4) * PW + (x1c + 4)) * 64;
        const _Float16* p10 = xhp + ((size_t)(y1c + 4) * PW + (x0c + 4)) * 64;
        const _Float16* p11 = xhp + ((size_t)(y1c + 4) * PW + (x1c + 4)) * 64;
        _Float16* dst = &ldsS[pix * DSROW + k * 64];
#pragma unroll
        for (int cb = 0; cb < 8; ++cb) {
            h8 a0 = *(const h8*)(p00 + cb * 8);
            h8 a1 = *(const h8*)(p01 + cb * 8);
            h8 a2 = *(const h8*)(p10 + cb * 8);
            h8 a3 = *(const h8*)(p11 + cb * 8);
            h8 r;
#pragma unroll
            for (int i = 0; i < 8; ++i)
                r[i] = (_Float16)((float)a0[i] * wa0 + (float)a1[i] * wc0 +
                                  (float)a2[i] * wa1 + (float)a3[i] * wc1);
            *(h8*)(dst + cb * 8) = r;
        }
    }
    __syncthreads();

    int lane = tid & 31, og = tid >> 5;
    int m    = lane & 15, s = lane >> 4;
    f8 acc = {};
    const _Float16* wbase = dwB + br * 18 * 2048 + (og * 16 + m) * 32 + s * 8;
#pragma unroll 2
    for (int chunk = 0; chunk < 18; ++chunk) {
        const _Float16* aP = &ldsS[m * DSROW + chunk * 32 + s * 8];
        h16 a  = hcat(*(const h8*)aP, *(const h8*)(aP + 16));
        const _Float16* wq = wbase + chunk * 2048;
        h16 bm = hcat(*(const h8*)wq, *(const h8*)(wq + 16));
        acc = __builtin_amdgcn_wmma_f32_16x16x32_f16(
                  false, a, false, bm, (short)0, acc, false, false);
    }
    float bias = (br ? dby : dbx)[og * 16 + m];
#pragma unroll
    for (int r = 0; r < 8; ++r)
        ldsC[(og * 16 + m) * 16 + (r + 8 * s)] = acc[r] + bias;
    __syncthreads();
    int oc = tid & 63, half = tid >> 6;
    float* op = preGN + ((((br * 2 + b) * 64 + oc) << 14) + (w << 7) + h0 + half * 8);
#pragma unroll
    for (int i = 0; i < 8; ++i)
        op[i] = ldsC[oc * 16 + half * 8 + i];
}

// ---------------------------------------------------------------------------
// GroupNorm stats: one block per (br,b,group) — deterministic tree reduction.
// ---------------------------------------------------------------------------
__global__ __launch_bounds__(256) void kstats(const float* __restrict__ preGN,
                                              float* __restrict__ gstats) {
    __shared__ float ssum[256], ssq[256];
    int bid = blockIdx.x;                            // (br*2+b)*16 + g, 64 blocks
    int t   = threadIdx.x;
    const float* base = preGN + ((size_t)((bid >> 4) * 64 + (bid & 15) * 4) << 14);
    float a = 0.f, q = 0.f;
    for (int j = t; j < 65536; j += 256) {
        float v = base[j];
        a += v; q += v * v;
    }
    ssum[t] = a; ssq[t] = q;
    __syncthreads();
    for (int st = 128; st > 0; st >>= 1) {
        if (t < st) { ssum[t] += ssum[t + st]; ssq[t] += ssq[t + st]; }
        __syncthreads();
    }
    if (t == 0) { gstats[bid * 2] = ssum[0]; gstats[bid * 2 + 1] = ssq[0]; }
}

// ---------------------------------------------------------------------------
// Apply GroupNorm + ReLU, write to output channels 64..191.
// ---------------------------------------------------------------------------
__global__ __launch_bounds__(256) void kgn(const float* __restrict__ preGN,
                                           const float* __restrict__ gstats,
                                           const float* __restrict__ gxg,
                                           const float* __restrict__ gxb,
                                           const float* __restrict__ gyg,
                                           const float* __restrict__ gyb,
                                           float* __restrict__ out) {
    int idx = blockIdx.x * 256 + threadIdx.x;        // 2*2*64*16384
    if (idx >= 2 * 2 * 64 * 16384) return;
    int sp = idx & 16383;
    int oc = (idx >> 14) & 63;
    int b  = (idx >> 20) & 1;
    int br = idx >> 21;
    int sb = ((br * 2 + b) * 16 + (oc >> 2)) * 2;
    const float NINV = 1.f / 65536.f;
    float mean = gstats[sb] * NINV;
    float var  = gstats[sb + 1] * NINV - mean * mean;
    float inv  = rsqrtf(var + 1e-5f);
    float gg = (br ? gyg : gxg)[oc];
    float bb = (br ? gyb : gxb)[oc];
    float v  = (preGN[idx] - mean) * inv * gg + bb;
    out[(((b * 192 + 64 + br * 64 + oc) << 14) + sp)] = fmaxf(v, 0.f);
}

// ---------------------------------------------------------------------------
extern "C" void kernel_launch(void* const* d_in, const int* in_sizes, int n_in,
                              void* d_out, int out_size, void* d_ws, size_t ws_size,
                              hipStream_t stream) {
    const float* x        = (const float*)d_in[0];
    const float* conv0_w  = (const float*)d_in[1];
    const float* conv0_g  = (const float*)d_in[2];
    const float* conv0_b  = (const float*)d_in[3];
    const float* conv0_m  = (const float*)d_in[4];
    const float* conv0_v  = (const float*)d_in[5];
    const float* offx_w   = (const float*)d_in[6];
    const float* offx_b   = (const float*)d_in[7];
    const float* offx_bng = (const float*)d_in[8];
    const float* offx_bnb = (const float*)d_in[9];
    const float* offx_bnm = (const float*)d_in[10];
    const float* offx_bnv = (const float*)d_in[11];
    const float* dscx_w   = (const float*)d_in[12];
    const float* dscx_b   = (const float*)d_in[13];
    const float* gnx_g    = (const float*)d_in[14];
    const float* gnx_b    = (const float*)d_in[15];
    const float* offy_w   = (const float*)d_in[16];
    const float* offy_b   = (const float*)d_in[17];
    const float* offy_bng = (const float*)d_in[18];
    const float* offy_bnb = (const float*)d_in[19];
    const float* offy_bnm = (const float*)d_in[20];
    const float* offy_bnv = (const float*)d_in[21];
    const float* dscy_w   = (const float*)d_in[22];
    const float* dscy_b   = (const float*)d_in[23];
    const float* gny_g    = (const float*)d_in[24];
    const float* gny_b    = (const float*)d_in[25];
    float* out = (float*)d_out;

    char* ws = (char*)d_ws;
    _Float16* xhp   = (_Float16*)(ws + 0);                 //  4,734,976 B (2*136*136*64*2)
    _Float16* w0B   = (_Float16*)(ws + 4734976);           //    663,552 B
    _Float16* dwB   = (_Float16*)(ws + 5398528);           //    147,456 B
    _Float16* owB   = (_Float16*)(ws + 5545984);           //     36,864 B
    float*    offb  = (float*)   (ws + 5582848);           //  2,359,296 B
    float*    preGN = (float*)   (ws + 7942144);           // 16,777,216 B
    float*    gstat = (float*)   (ws + 24719360);          //        512 B

    kpackxpad<<<9248, 256, 0, stream>>>(x, xhp);
    kpackw0  <<<1296, 256, 0, stream>>>(conv0_w, w0B);
    kpackdw  <<<288,  256, 0, stream>>>(dscx_w, dscy_w, dwB);
    kpackow  <<<72,   256, 0, stream>>>(offx_w, offy_w, owB);

    kconv0   <<<2048, 128, 0, stream>>>(xhp, w0B, conv0_g, conv0_b, conv0_m, conv0_v, out);

    koffw    <<<2048, 128, 0, stream>>>(xhp, owB,
        offx_b, offx_bng, offx_bnb, offx_bnm, offx_bnv,
        offy_b, offy_bng, offy_bnb, offy_bnm, offy_bnv, offb);

    kdsc     <<<4096, 128, 0, stream>>>(xhp, offb, dwB, dscx_b, dscy_b, preGN);
    kstats   <<<64,   256, 0, stream>>>(preGN, gstat);
    kgn      <<<16384,256, 0, stream>>>(preGN, gstat, gnx_g, gnx_b, gny_g, gny_b, out);
}
